// GPSnet_42245298323958
// MI455X (gfx1250) — matrix operand order repaired and checked by reading, not compiled, not run on hardware
//
#include <hip/hip_runtime.h>
#include <math.h>

typedef __attribute__((ext_vector_type(16))) _Float16 v16h;
typedef __attribute__((ext_vector_type(4)))  _Float16 v4h;
typedef __attribute__((ext_vector_type(8)))  float    v8f;

#define Nn   32768
#define Ee   524288
#define Cc   128
#define Ll   3
#define Hh   4
#define Bb   64
#define Ss   512
#define HIDD 256

// ---------------------------------------------------------------------------
// helpers
// ---------------------------------------------------------------------------
__device__ __forceinline__ void atomAddF(float* p, float v) {
  __hip_atomic_fetch_add(p, v, __ATOMIC_RELAXED, __HIP_MEMORY_SCOPE_AGENT);
}

__device__ __forceinline__ float gelu_erf(float x) {
  return 0.5f * x * (1.0f + erff(x * 0.70710678118654752f));
}

__device__ __forceinline__ v4h cvt4(float4 v) {
  v4h h;
  h[0] = (_Float16)v.x; h[1] = (_Float16)v.y;
  h[2] = (_Float16)v.z; h[3] = (_Float16)v.w;
  return h;
}

#define WMMA_F16(a, b, c) \
  __builtin_amdgcn_wmma_f32_16x16x32_f16(false, (a), false, (b), (short)0, (c), false, false)

// ---------------------------------------------------------------------------
// fast WMMA GEMM:  D[M,N] = act( A[M,K] @ B[K,N] (+bias) (+res) )
// Preconditions: K % 32 == 0, M % 128 == 0, N % 64 == 0, 16B-aligned rows.
// block = 128 threads (4 waves); block tile 128x64; wave tile 32x64
// per k-step: 6 fragment loads -> 8 back-to-back WMMAs (B frags reused x2)
// ---------------------------------------------------------------------------
__global__ __launch_bounds__(128) void k_gemm(
    const float* __restrict__ A, int lda,
    const float* __restrict__ Bm,
    const float* __restrict__ bias, const float* __restrict__ res,
    float* __restrict__ D, int N, int K, int act)
{
  __shared__ _Float16 sA[128][40];   // 32 used cols + pad (conflict-free frag reads)
  __shared__ _Float16 sB[32][64];

  const int wave = threadIdx.x >> 5;
  const int lane = threadIdx.x & 31;
  const int halfq = lane >> 4;
  const int l16 = lane & 15;
  const int kb = halfq * 8;                 // A-frag K sub-base
  const int krow = l16 + 16 * halfq;        // B-frag K row

  // staging: A -> one row per thread; B -> 4 threads per k-row
  const int bk_ld = threadIdx.x >> 2;             // B k-row 0..31
  const int bn_ld = (threadIdx.x & 3) * 16;       // B col base {0,16,32,48}

  const float* Arow = A + (size_t)(blockIdx.y * 128 + threadIdx.x) * lda;
  const float* Brow = Bm + (size_t)bk_ld * N + blockIdx.x * 64 + bn_ld;

  v8f acc[2][4];
#pragma unroll
  for (int s = 0; s < 2; ++s)
#pragma unroll
    for (int t = 0; t < 4; ++t)
#pragma unroll
      for (int i = 0; i < 8; ++i) acc[s][t][i] = 0.0f;

  for (int k0 = 0; k0 < K; k0 += 32) {
    // ---- stage A (128x32) : 8 float4 per thread, one row ----
    const float4* ap = (const float4*)(Arow + k0);
#pragma unroll
    for (int j = 0; j < 8; ++j)
      *(v4h*)&sA[threadIdx.x][j * 4] = cvt4(ap[j]);
    // ---- stage B (32x64) : 4 float4 per thread ----
    const float4* bp = (const float4*)(Brow + (size_t)k0 * N);
#pragma unroll
    for (int j = 0; j < 4; ++j)
      *(v4h*)&sB[bk_ld][bn_ld + j * 4] = cvt4(bp[j]);
    if (k0 + 32 < K)
      __builtin_prefetch(Arow + k0 + 32, 0, 1);   // global_prefetch_b8
    __syncthreads();

    // ---- load ALL fragments first (one ds clause), then WMMAs back-to-back
    v16h af0, af1;
    const int ar0 = wave * 32 + l16;
#pragma unroll
    for (int i = 0; i < 8; ++i) {
      af0[i]     = sA[ar0][kb + i];
      af0[8 + i] = sA[ar0][kb + 16 + i];
      af1[i]     = sA[ar0 + 16][kb + i];
      af1[8 + i] = sA[ar0 + 16][kb + 16 + i];
    }
    v16h bf[4];
#pragma unroll
    for (int t = 0; t < 4; ++t)
#pragma unroll
      for (int e = 0; e < 16; ++e) bf[t][e] = sB[e < 8 ? kb + e : kb + 8 + e][t * 16 + l16];

#pragma unroll
    for (int t = 0; t < 4; ++t) {
      acc[0][t] = WMMA_F16(af0, bf[t], acc[0][t]);
      acc[1][t] = WMMA_F16(af1, bf[t], acc[1][t]);
    }
    __syncthreads();
  }

  // ---- epilogue ----
#pragma unroll
  for (int s = 0; s < 2; ++s)
#pragma unroll
    for (int t = 0; t < 4; ++t)
#pragma unroll
      for (int i = 0; i < 8; ++i) {
        int gr = blockIdx.y * 128 + wave * 32 + s * 16 + halfq * 8 + i;
        int gc = blockIdx.x * 64 + t * 16 + l16;
        float v = acc[s][t][i];
        if (bias) v += bias[gc];
        if (res)  v += res[(size_t)gr * N + gc];
        if (act == 1) v = fmaxf(v, 0.0f);
        D[(size_t)gr * N + gc] = v;
      }
}

// ---------------------------------------------------------------------------
// flash attention per (graph, head):  S=512, HD=32, scale 1/sqrt(32)
// block = 128 threads = 4 waves; block handles 64 queries (wave: 16)
// grid.x = B*H*(S/64) = 2048
// ---------------------------------------------------------------------------
__global__ __launch_bounds__(128) void k_attn(const float* __restrict__ qkv,
                                              float* __restrict__ ao)
{
  __shared__ _Float16 sKT[32][512];     // K^T : [hd][key]   32 KB
  __shared__ _Float16 sV[512][32];      // V   : [key][hd]   32 KB
  __shared__ _Float16 sP[4][16][40];    // per-wave prob tile 16x32 (+pad)

  const int qblk = blockIdx.x & 7;
  const int bh = blockIdx.x >> 3;
  const int hh = bh & 3;
  const int bb = bh >> 2;
  const int tid = threadIdx.x;

  // stage K^T and V (float4 loads)
  for (int idx = tid; idx < 512 * 8; idx += 128) {
    int s = idx >> 3, dq = (idx & 7) * 4;
    const float* base = qkv + ((size_t)(bb * 512 + s)) * 384 + hh * 32 + dq;
    float4 kv = *(const float4*)(base + 128);
    float4 vv = *(const float4*)(base + 256);
    sKT[dq + 0][s] = (_Float16)kv.x;
    sKT[dq + 1][s] = (_Float16)kv.y;
    sKT[dq + 2][s] = (_Float16)kv.z;
    sKT[dq + 3][s] = (_Float16)kv.w;
    *(v4h*)&sV[s][dq] = cvt4(vv);
  }
  __syncthreads();

  const int wave = tid >> 5, lane = tid & 31;
  const int halfq = lane >> 4, l16 = lane & 15;
  const int kb = halfq * 8;
  const int krow = l16 + 16 * halfq;
  const int q0 = qblk * 64 + wave * 16;

  // Q fragment (A-matrix 16x32) straight from global
  v16h af;
  {
    const float* qrow = qkv + ((size_t)(bb * 512 + q0 + l16)) * 384 + hh * 32;
#pragma unroll
    for (int i = 0; i < 8; ++i) {
      af[i]     = (_Float16)qrow[kb + i];
      af[8 + i] = (_Float16)qrow[kb + 16 + i];
    }
  }

  const float scale = 0.17677669529663687f;   // 1/sqrt(32)
  float run_max[8], run_sum[8];
  v8f acc0, acc1, zc;
#pragma unroll
  for (int i = 0; i < 8; ++i) {
    run_max[i] = -1e30f; run_sum[i] = 0.0f;
    acc0[i] = 0.0f; acc1[i] = 0.0f; zc[i] = 0.0f;
  }

  for (int kt = 0; kt < 512; kt += 32) {
    // two 16x16 score tiles (K = HD = 32, one WMMA each)
    v16h bf0, bf1;
#pragma unroll
    for (int e = 0; e < 16; ++e) {
      bf0[e] = sKT[e < 8 ? kb + e : kb + 8 + e][kt + l16];
      bf1[e] = sKT[e < 8 ? kb + e : kb + 8 + e][kt + 16 + l16];
    }
    v8f s0 = WMMA_F16(af, bf0, zc);
    v8f s1 = WMMA_F16(af, bf1, zc);

    // online softmax update (row = i + 8*halfq, stats uniform per 16-lane half)
#pragma unroll
    for (int i = 0; i < 8; ++i) {
      float a = s0[i] * scale, b = s1[i] * scale;
      float m = fmaxf(a, b);
      m = fmaxf(m, __shfl_xor(m, 1, 16));
      m = fmaxf(m, __shfl_xor(m, 2, 16));
      m = fmaxf(m, __shfl_xor(m, 4, 16));
      m = fmaxf(m, __shfl_xor(m, 8, 16));
      float nm = fmaxf(run_max[i], m);
      float f = __expf(run_max[i] - nm);
      run_max[i] = nm;
      acc0[i] *= f; acc1[i] *= f;
      float p0 = __expf(a - nm), p1 = __expf(b - nm);
      float rs = p0 + p1;
      rs += __shfl_xor(rs, 1, 16);
      rs += __shfl_xor(rs, 2, 16);
      rs += __shfl_xor(rs, 4, 16);
      rs += __shfl_xor(rs, 8, 16);
      run_sum[i] = run_sum[i] * f + rs;
      int r = i + 8 * halfq;
      sP[wave][r][l16]      = (_Float16)p0;
      sP[wave][r][16 + l16] = (_Float16)p1;
    }
    __syncthreads();

    // P (A 16x32) @ V tile (32x32) -> 2 accum tiles
    v16h pf;
#pragma unroll
    for (int i = 0; i < 8; ++i) { pf[i] = sP[wave][l16][kb + i]; pf[8 + i] = sP[wave][l16][kb + 16 + i]; }
    v16h vf0, vf1;
#pragma unroll
    for (int e = 0; e < 16; ++e) {
      vf0[e] = sV[kt + (e < 8 ? kb + e : kb + 8 + e)][l16];
      vf1[e] = sV[kt + (e < 8 ? kb + e : kb + 8 + e)][16 + l16];
    }
    acc0 = WMMA_F16(pf, vf0, acc0);
    acc1 = WMMA_F16(pf, vf1, acc1);
    __syncthreads();
  }

#pragma unroll
  for (int i = 0; i < 8; ++i) {
    float inv = 1.0f / run_sum[i];
    size_t node = (size_t)bb * 512 + q0 + i + 8 * halfq;
    ao[node * 128 + hh * 32 + l16]      = acc0[i] * inv;
    ao[node * 128 + hh * 32 + 16 + l16] = acc1[i] * inv;
  }
}

// ---------------------------------------------------------------------------
// small kernels
// ---------------------------------------------------------------------------
__global__ void k_fill1(float* p, int n) {
  int i = blockIdx.x * blockDim.x + threadIdx.x;
  if (i < n) p[i] = 1.0f;
}
__global__ void k_deg_scatter(const int* __restrict__ ei, float* __restrict__ deg) {
  int e = blockIdx.x * blockDim.x + threadIdx.x;
  if (e < Ee) atomAddF(&deg[ei[Ee + e]], 1.0f);
}
__global__ void k_rsqrt_ip(float* p, int n) {
  int i = blockIdx.x * blockDim.x + threadIdx.x;
  if (i < n) p[i] = rsqrtf(p[i]);
}
// transpose W[rows][cols] -> WT[cols][rows]
__global__ void k_transpose(const float* __restrict__ W, float* __restrict__ WT,
                            int rows, int cols) {
  int idx = blockIdx.x * blockDim.x + threadIdx.x;
  if (idx < rows * cols) {
    int r = idx / cols, c = idx % cols;
    WT[(size_t)c * rows + r] = W[idx];
  }
}
// pack [x | pe | 0] -> Apad[Nn][96] fp32
__global__ void k_pack_embed(const float* __restrict__ x, const float* __restrict__ pe,
                             float* __restrict__ Apad) {
  int idx = blockIdx.x * blockDim.x + threadIdx.x;
  if (idx < Nn * 96) {
    int n = idx / 96, k = idx - n * 96;
    float v = 0.0f;
    if (k < 64)      v = x[n * 64 + k];
    else if (k < 80) v = pe[n * 16 + (k - 64)];
    Apad[idx] = v;
  }
}
// loc init: self-loop term  loc = xw * dinv^2
__global__ void k_loc_init(const float* __restrict__ xw, const float* __restrict__ dinv,
                           float* __restrict__ loc) {
  int idx = blockIdx.x * blockDim.x + threadIdx.x;
  if (idx < Nn * Cc) {
    int i = idx >> 7;
    float di = dinv[i];
    loc[idx] = xw[idx] * di * di;
  }
}
// edge scatter: loc[dst] += xw[src] * dinv[src]*dinv[dst], 4 channels / thread
__global__ void k_edge_scatter(const int* __restrict__ ei, const float* __restrict__ dinv,
                               const float* __restrict__ xw, float* __restrict__ loc) {
  long long t = (long long)blockIdx.x * blockDim.x + threadIdx.x;
  if (t >= (long long)Ee * 32) return;
  int e = (int)(t >> 5);
  int ch = ((int)t & 31) * 4;
  int s = ei[e], d = ei[Ee + e];
  float cf = dinv[s] * dinv[d];
  const float4 xs = *(const float4*)(xw + (size_t)s * Cc + ch);
  float* ld = loc + (size_t)d * Cc + ch;
  atomAddF(&ld[0], xs.x * cf);
  atomAddF(&ld[1], xs.y * cf);
  atomAddF(&ld[2], xs.z * cf);
  atomAddF(&ld[3], xs.w * cf);
}
// Y = A (+B) (+bias[c])
__global__ void k_ew_add(const float* __restrict__ A, const float* __restrict__ Bp,
                         const float* __restrict__ bias, float* __restrict__ Y, int total) {
  int idx = blockIdx.x * blockDim.x + threadIdx.x;
  if (idx < total) {
    float v = A[idx];
    if (Bp)   v += Bp[idx];
    if (bias) v += bias[idx & (Cc - 1)];
    Y[idx] = v;
  }
}
// per-channel mean / rsqrt(var+eps) over Nn rows
__global__ __launch_bounds__(256) void k_bn_stats(const float* __restrict__ X,
                                                  float* __restrict__ mu, float* __restrict__ rs) {
  __shared__ float r1[256], r2[256];
  int c = blockIdx.x;
  float a = 0.0f, b = 0.0f;
  for (int r = threadIdx.x; r < Nn; r += 256) {
    float v = X[(size_t)r * Cc + c];
    a += v; b += v * v;
  }
  r1[threadIdx.x] = a; r2[threadIdx.x] = b;
  __syncthreads();
  for (int st = 128; st > 0; st >>= 1) {
    if (threadIdx.x < st) { r1[threadIdx.x] += r1[threadIdx.x + st]; r2[threadIdx.x] += r2[threadIdx.x + st]; }
    __syncthreads();
  }
  if (threadIdx.x == 0) {
    float m = r1[0] * (1.0f / Nn);
    float v = r2[0] * (1.0f / Nn) - m * m;
    mu[c] = m;
    rs[c] = rsqrtf(v + 1e-5f);
  }
}
__global__ void k_bn_apply(const float* __restrict__ X, const float* __restrict__ mu,
                           const float* __restrict__ rs, const float* __restrict__ g,
                           const float* __restrict__ be, float* __restrict__ Y,
                           int total, int dogelu) {
  int idx = blockIdx.x * blockDim.x + threadIdx.x;
  if (idx < total) {
    int c = idx & (Cc - 1);
    float v = (X[idx] - mu[c]) * rs[c] * g[c] + be[c];
    if (dogelu) v = gelu_erf(v);
    Y[idx] = v;
  }
}
__global__ void k_pool(const float* __restrict__ h, float* __restrict__ pooled) {
  int b = blockIdx.x, c = threadIdx.x;   // 64 blocks x 128 threads
  float s = 0.0f;
  for (int i = 0; i < Ss; ++i) s += h[((size_t)b * Ss + i) * Cc + c];
  pooled[b * Cc + c] = s * (1.0f / Ss);
}
__global__ void k_head1(const float* __restrict__ pooled, const float* __restrict__ W,
                        const float* __restrict__ bi, float* __restrict__ z) {
  int idx = blockIdx.x * blockDim.x + threadIdx.x;   // 64*256
  if (idx < Bb * HIDD) {
    int b = idx >> 8, j = idx & 255;
    float s = bi[j];
    for (int k = 0; k < Cc; ++k) s += pooled[b * Cc + k] * W[k * HIDD + j];
    z[idx] = gelu_erf(s);
  }
}
__global__ void k_head2(const float* __restrict__ z, const float* __restrict__ W,
                        const float* __restrict__ bi, float* __restrict__ out) {
  int b = threadIdx.x;
  if (b < Bb) {
    float s = bi[0];
    for (int k = 0; k < HIDD; ++k) s += z[b * HIDD + k] * W[k];
    out[b] = s;
  }
}

// ---------------------------------------------------------------------------
// host orchestration
// ---------------------------------------------------------------------------
extern "C" void kernel_launch(void* const* d_in, const int* in_sizes, int n_in,
                              void* d_out, int out_size, void* d_ws, size_t ws_size,
                              hipStream_t stream) {
  const float* x      = (const float*)d_in[0];
  const float* pe     = (const float*)d_in[1];
  const int*   ei     = (const int*)d_in[2];
  const float* W_emb  = (const float*)d_in[4];
  const float* b_emb  = (const float*)d_in[5];
  const float* gcn_w  = (const float*)d_in[6];
  const float* gcn_b  = (const float*)d_in[7];
  const float* in_w   = (const float*)d_in[8];
  const float* in_b   = (const float*)d_in[9];
  const float* out_w  = (const float*)d_in[10];
  const float* out_b  = (const float*)d_in[11];
  const float* mlp_w1 = (const float*)d_in[12];
  const float* mlp_b1 = (const float*)d_in[13];
  const float* mlp_w2 = (const float*)d_in[14];
  const float* mlp_b2 = (const float*)d_in[15];
  const float* g1  = (const float*)d_in[16];
  const float* be1 = (const float*)d_in[17];
  const float* g2  = (const float*)d_in[18];
  const float* be2 = (const float*)d_in[19];
  const float* g3  = (const float*)d_in[20];
  const float* be3 = (const float*)d_in[21];
  const float* W_hid = (const float*)d_in[22];
  const float* b_hid = (const float*)d_in[23];
  const float* W_out = (const float*)d_in[24];
  const float* b_out = (const float*)d_in[25];
  float* out = (float*)d_out;

  // workspace carve-up (floats)
  float* ws = (float*)d_ws;
  size_t off = 0;
  float* hbuf = ws + off; off += (size_t)Nn * Cc;
  float* bufA = ws + off; off += (size_t)Nn * Cc;
  float* bufB = ws + off; off += (size_t)Nn * Cc;
  float* bufC = ws + off; off += (size_t)Nn * Cc;
  float* bufQ = ws + off; off += (size_t)Nn * 3 * Cc;
  float* dinv = ws + off; off += Nn;
  float* mu   = ws + off; off += Cc;
  float* rs   = ws + off; off += Cc;
  float* pooled = ws + off; off += (size_t)Bb * Cc;
  float* zbuf   = ws + off; off += (size_t)Bb * HIDD;
  float* in_wT  = ws + off; off += (size_t)Ll * Cc * 3 * Cc;   // [l][128][384]
  float* out_wT = ws + off; off += (size_t)Ll * Cc * Cc;       // [l][128][128]

  const int NC = Nn * Cc;
  const dim3 ew_grid((NC + 255) / 256);

  // --- weight pre-transposes (so every GEMM is B[K,N]) ---
  for (int l = 0; l < Ll; ++l) {
    k_transpose<<<(3 * Cc * Cc + 255) / 256, 256, 0, stream>>>(
        in_w + (size_t)l * 3 * Cc * Cc, in_wT + (size_t)l * Cc * 3 * Cc, 3 * Cc, Cc);
    k_transpose<<<(Cc * Cc + 255) / 256, 256, 0, stream>>>(
        out_w + (size_t)l * Cc * Cc, out_wT + (size_t)l * Cc * Cc, Cc, Cc);
  }

  // --- GCN normalization coefficients ---
  k_fill1<<<Nn / 256, 256, 0, stream>>>(dinv, Nn);
  k_deg_scatter<<<Ee / 256, 256, 0, stream>>>(ei, dinv);
  k_rsqrt_ip<<<Nn / 256, 256, 0, stream>>>(dinv, Nn);

  // --- embedding: h = [x|pe|0][N,96] @ Wpad[96,128] + b_emb ---
  k_pack_embed<<<(Nn * 96 + 255) / 256, 256, 0, stream>>>(x, pe, bufQ);
  {
    float* Wpad = bufC;   // scratch head, reused later
    hipMemsetAsync(Wpad, 0, 96 * Cc * sizeof(float), stream);
    hipMemcpyAsync(Wpad, W_emb, 80 * Cc * sizeof(float),
                   hipMemcpyDeviceToDevice, stream);
    k_gemm<<<dim3(Cc / 64, Nn / 128), 128, 0, stream>>>(
        bufQ, 96, Wpad, b_emb, nullptr, hbuf, Cc, 96, 0);
  }

  for (int l = 0; l < Ll; ++l) {
    const float* gcn_w_l  = gcn_w  + (size_t)l * Cc * Cc;
    const float* gcn_b_l  = gcn_b  + (size_t)l * Cc;
    const float* in_wT_l  = in_wT  + (size_t)l * Cc * 3 * Cc;
    const float* in_b_l   = in_b   + (size_t)l * 3 * Cc;
    const float* out_wT_l = out_wT + (size_t)l * Cc * Cc;
    const float* out_b_l  = out_b  + (size_t)l * Cc;
    const float* mlp_w1_l = mlp_w1 + (size_t)l * Cc * 2 * Cc;
    const float* mlp_b1_l = mlp_b1 + (size_t)l * 2 * Cc;
    const float* mlp_w2_l = mlp_w2 + (size_t)l * 2 * Cc * Cc;
    const float* mlp_b2_l = mlp_b2 + (size_t)l * Cc;

    // ---- local GCN branch ----
    k_gemm<<<dim3(Cc / 64, Nn / 128), 128, 0, stream>>>(
        hbuf, Cc, gcn_w_l, nullptr, nullptr, bufA, Cc, Cc, 0);
    k_loc_init<<<ew_grid, 256, 0, stream>>>(bufA, dinv, bufB);
    k_edge_scatter<<<(unsigned)(((long long)Ee * 32 + 255) / 256), 256, 0, stream>>>(
        ei, dinv, bufA, bufB);
    k_ew_add<<<ew_grid, 256, 0, stream>>>(bufB, hbuf, gcn_b_l, bufB, NC);   // pre1
    k_bn_stats<<<Cc, 256, 0, stream>>>(bufB, mu, rs);
    k_bn_apply<<<ew_grid, 256, 0, stream>>>(bufB, mu, rs, g1 + l * Cc, be1 + l * Cc,
                                            bufB, NC, 0);                    // h1

    // ---- global attention branch ----
    k_gemm<<<dim3(3 * Cc / 64, Nn / 128), 128, 0, stream>>>(
        hbuf, Cc, in_wT_l, in_b_l, nullptr, bufQ, 3 * Cc, Cc, 0);
    k_attn<<<Bb * Hh * (Ss / 64), 128, 0, stream>>>(bufQ, bufA);             // ao
    k_gemm<<<dim3(Cc / 64, Nn / 128), 128, 0, stream>>>(
        bufA, Cc, out_wT_l, out_b_l, hbuf, bufC, Cc, Cc, 0);                 // pre2
    k_bn_stats<<<Cc, 256, 0, stream>>>(bufC, mu, rs);
    k_bn_apply<<<ew_grid, 256, 0, stream>>>(bufC, mu, rs, g2 + l * Cc, be2 + l * Cc,
                                            bufC, NC, 0);                    // h2

    // ---- combine + MLP ----
    k_ew_add<<<ew_grid, 256, 0, stream>>>(bufB, bufC, nullptr, bufB, NC);    // out = h1+h2
    k_gemm<<<dim3(2 * Cc / 64, Nn / 128), 128, 0, stream>>>(
        bufB, Cc, mlp_w1_l, mlp_b1_l, nullptr, bufQ, 2 * Cc, Cc, 1);
    k_gemm<<<dim3(Cc / 64, Nn / 128), 128, 0, stream>>>(
        bufQ, 2 * Cc, mlp_w2_l, mlp_b2_l, bufB, bufC, Cc, 2 * Cc, 0);        // pre3
    k_bn_stats<<<Cc, 256, 0, stream>>>(bufC, mu, rs);
    k_bn_apply<<<ew_grid, 256, 0, stream>>>(bufC, mu, rs, g3 + l * Cc, be3 + l * Cc,
                                            hbuf, NC, 1);                    // h = gelu(bn)
  }

  // ---- readout ----
  k_pool<<<Bb, Cc, 0, stream>>>(hbuf, pooled);
  k_head1<<<(Bb * HIDD) / 256, 256, 0, stream>>>(pooled, W_hid, b_hid, zbuf);
  k_head2<<<1, 64, 0, stream>>>(zbuf, W_out, b_out, out);
}
